// MiniBitterLLM_59365037965388
// MI455X (gfx1250) — matrix operand order repaired and checked by reading, not compiled, not run on hardware
//
#include <hip/hip_runtime.h>
#include <hip/hip_bf16.h>

// ---------------------------------------------------------------------------
// MiniBitterLLM forward for MI455X (gfx1250, wave32).
// GEMMs: bf16 WMMA (v_wmma_f32_16x16x32_bf16), f32 accumulate,
//        one wave computes a 16x64 output strip (4 acc tiles share 1 A frag),
//        ping-pong double-buffered over K (no register rotation copies),
//        global_prefetch_b8 one step ahead.  Requires K % 64 == 0 (holds for
//        all GEMMs here: K = 512 or 2048).
// Attention: wave32-per-query windowed online-softmax (tiny FLOPs).
// ---------------------------------------------------------------------------

#define Bc 2
#define Sc 1024
#define Dc 512
#define Hc 8
#define DHc 64
#define Vc 32000
#define Mc (Bc * Sc)          // 2048 rows
#define RATEc 0.25f
#define LNEPS 1e-5f
#define NEG_BIG (-3.402823466e38f)

#define CDIV(a, b) (((a) + (b) - 1) / (b))

typedef __attribute__((ext_vector_type(16))) __bf16 v16bf;
typedef __attribute__((ext_vector_type(8)))  float  v8f;

// ---------------- f32 -> bf16 (RNE) conversion ----------------
__global__ __launch_bounds__(256) void k_f32_to_bf16(const float* __restrict__ src,
                                                     unsigned short* __restrict__ dst,
                                                     size_t n) {
  size_t i = (size_t)blockIdx.x * blockDim.x + threadIdx.x;
  if (i >= n) return;
  union { float f; unsigned u; } v; v.f = src[i];
  unsigned b = v.u + 0x7FFFu + ((v.u >> 16) & 1u);
  dst[i] = (unsigned short)(b >> 16);
}

// ---------------- fragment loaders (contiguous k order, same for A and B) ----------------
//  A (16x32 bf16): lanes 0-15 rows M=0..15, K=0..15; lanes 16-31
//    same rows, K=16..31; 2 bf16 per VGPR.
//  B (32x16 bf16): lane = column N, lanes 0-15 K=0..15, lanes 16-31 K=16..31.
__device__ inline v16bf load_afrag(const unsigned short* __restrict__ arow,
                                   int k0, int half) {
  union { v16bf v; unsigned short u[16]; } f;
#pragma unroll
  for (int j = 0; j < 8; ++j) {
    int kk = (half << 4) + (j << 1);
    f.u[2 * j]     = arow[k0 + kk];
    f.u[2 * j + 1] = arow[k0 + kk + 1];
  }
  return f.v;
}

__device__ inline v16bf load_bfrag(const unsigned short* __restrict__ brow, int k0) {
  union { v16bf v; unsigned short u[16]; } f;
#pragma unroll
  for (int j = 0; j < 8; ++j) {
    f.u[2 * j]     = brow[k0 + 2 * j];
    f.u[2 * j + 1] = brow[k0 + 2 * j + 1];
  }
  return f.v;
}

// ---------------- WMMA GEMM: C[M,N] = A[M,K] * W[N,K]^T + bias ----------------
// One wave computes a 16x64 strip (4 x 16x16 tiles), ping-pong over K.
// C/D f32 layout: VGPR v -> row v + 8*(lane/16), col = lane%16.
__global__ __launch_bounds__(32) void k_gemm_bf16(const unsigned short* __restrict__ A,
                                                  const unsigned short* __restrict__ W,
                                                  const float* __restrict__ bias,
                                                  float* __restrict__ C,
                                                  int M, int N, int K, int relu) {
  const int tileM = blockIdx.y;
  const int nBase = blockIdx.x * 64;   // 4 consecutive 16-wide N tiles
  const int lane = threadIdx.x;
  const int half = lane >> 4;
  const int l16  = lane & 15;

  v8f acc0 = {}, acc1 = {}, acc2 = {}, acc3 = {};
  const int rowA = tileM * 16 + l16;
  const unsigned short* arow  = A + (size_t)rowA * K;
  const unsigned short* brow0 = W + (size_t)(nBase +  0 + l16) * K + (half << 4);
  const unsigned short* brow1 = W + (size_t)(nBase + 16 + l16) * K + (half << 4);
  const unsigned short* brow2 = W + (size_t)(nBase + 32 + l16) * K + (half << 4);
  const unsigned short* brow3 = W + (size_t)(nBase + 48 + l16) * K + (half << 4);

  // prologue: phase-A fragments at k=0
  v16bf aA  = load_afrag(arow, 0, half);
  v16bf b0A = load_bfrag(brow0, 0);
  v16bf b1A = load_bfrag(brow1, 0);
  v16bf b2A = load_bfrag(brow2, 0);
  v16bf b3A = load_bfrag(brow3, 0);

  for (int k0 = 0; k0 < K; k0 += 64) {
    // phase-B loads (k0+32) issued before phase-A WMMAs consume their frags
    v16bf aB  = load_afrag(arow, k0 + 32, half);
    v16bf b0B = load_bfrag(brow0, k0 + 32);
    v16bf b1B = load_bfrag(brow1, k0 + 32);
    v16bf b2B = load_bfrag(brow2, k0 + 32);
    v16bf b3B = load_bfrag(brow3, k0 + 32);
    if (k0 + 64 < K) {
      // gfx1250 prefetch path (global_prefetch_b8), one 64-step ahead
      __builtin_prefetch(arow  + k0 + 64, 0, 0);
      __builtin_prefetch(brow0 + k0 + 64, 0, 0);
      __builtin_prefetch(brow1 + k0 + 64, 0, 0);
      __builtin_prefetch(brow2 + k0 + 64, 0, 0);
      __builtin_prefetch(brow3 + k0 + 64, 0, 0);
    }

    acc0 = __builtin_amdgcn_wmma_f32_16x16x32_bf16(false, aA, false, b0A,
                                                   (short)0, acc0, false, false);
    acc1 = __builtin_amdgcn_wmma_f32_16x16x32_bf16(false, aA, false, b1A,
                                                   (short)0, acc1, false, false);
    acc2 = __builtin_amdgcn_wmma_f32_16x16x32_bf16(false, aA, false, b2A,
                                                   (short)0, acc2, false, false);
    acc3 = __builtin_amdgcn_wmma_f32_16x16x32_bf16(false, aA, false, b3A,
                                                   (short)0, acc3, false, false);

    if (k0 + 64 < K) {
      // refill phase-A (k0+64) before phase-B WMMAs run
      aA  = load_afrag(arow, k0 + 64, half);
      b0A = load_bfrag(brow0, k0 + 64);
      b1A = load_bfrag(brow1, k0 + 64);
      b2A = load_bfrag(brow2, k0 + 64);
      b3A = load_bfrag(brow3, k0 + 64);
    }

    acc0 = __builtin_amdgcn_wmma_f32_16x16x32_bf16(false, aB, false, b0B,
                                                   (short)0, acc0, false, false);
    acc1 = __builtin_amdgcn_wmma_f32_16x16x32_bf16(false, aB, false, b1B,
                                                   (short)0, acc1, false, false);
    acc2 = __builtin_amdgcn_wmma_f32_16x16x32_bf16(false, aB, false, b2B,
                                                   (short)0, acc2, false, false);
    acc3 = __builtin_amdgcn_wmma_f32_16x16x32_bf16(false, aB, false, b3B,
                                                   (short)0, acc3, false, false);
  }

  const v8f accs[4] = {acc0, acc1, acc2, acc3};
#pragma unroll
  for (int sub = 0; sub < 4; ++sub) {
#pragma unroll
    for (int v = 0; v < 8; ++v) {
      int r = tileM * 16 + v + (half << 3);
      int c = nBase + sub * 16 + l16;
      if (r < M && c < N) {
        float val = accs[sub][v] + (bias ? bias[c] : 0.0f);
        if (relu) val = fmaxf(val, 0.0f);
        C[(size_t)r * N + c] = val;
      }
    }
  }
}

// ---------------- embedding gather ----------------
__global__ __launch_bounds__(256) void k_embed(const int* __restrict__ tokens,
                                               const float* __restrict__ emb,
                                               float* __restrict__ x, size_t total) {
  size_t i = (size_t)blockIdx.x * blockDim.x + threadIdx.x;
  if (i >= total) return;
  size_t row = i / Dc; int d = (int)(i % Dc);
  x[i] = emb[(size_t)tokens[row] * Dc + d];
}

// ---------------- windowed attention: one wave32 per (b,h,i) ----------------
__global__ __launch_bounds__(32) void k_attn(const float* __restrict__ qkv,
                                             float* __restrict__ out, int win) {
  int idx = blockIdx.x;
  int i = idx % Sc; idx /= Sc;
  int h = idx % Hc; idx /= Hc;
  int b = idx;
  int lane = threadIdx.x;
  const int D3 = 3 * Dc;
  const float scale = 0.125f;  // 1/sqrt(64)

  const float* qp = qkv + ((size_t)(b * Sc + i)) * D3 + h * DHc;
  float q0 = qp[2 * lane], q1 = qp[2 * lane + 1];
  float m = NEG_BIG, l = 0.f, o0 = 0.f, o1 = 0.f;
  int j0 = i - win + 1; if (j0 < 0) j0 = 0;
  for (int j = j0; j <= i; ++j) {
    const float* kp = qkv + ((size_t)(b * Sc + j)) * D3 + Dc + h * DHc;
    float p = q0 * kp[2 * lane] + q1 * kp[2 * lane + 1];
#pragma unroll
    for (int off = 16; off > 0; off >>= 1) p += __shfl_xor(p, off, 32);
    float s = p * scale;
    float mn = fmaxf(m, s);
    float corr = __expf(m - mn);
    float w = __expf(s - mn);
    const float* vp = qkv + ((size_t)(b * Sc + j)) * D3 + 2 * Dc + h * DHc;
    l  = l * corr + w;
    o0 = o0 * corr + w * vp[2 * lane];
    o1 = o1 * corr + w * vp[2 * lane + 1];
    m = mn;
  }
  float inv = 1.0f / l;
  float* op = out + ((size_t)(b * Sc + i)) * Dc + h * DHc;
  op[2 * lane]     = o0 * inv;
  op[2 * lane + 1] = o1 * inv;
}

// ---------------- residual + LayerNorm (block per row, D=512) ----------------
__global__ __launch_bounds__(256) void k_add_ln(const float* __restrict__ x,
                                                const float* __restrict__ r,
                                                const float* __restrict__ gs,
                                                const float* __restrict__ gb,
                                                float* __restrict__ outp) {
  __shared__ float sm[256];
  size_t base = (size_t)blockIdx.x * Dc;
  int t = threadIdx.x;
  float loc[2];
  float s = 0.f;
#pragma unroll
  for (int i = 0; i < 2; ++i) {
    float v = x[base + t + i * 256] + r[base + t + i * 256];
    loc[i] = v; s += v;
  }
  sm[t] = s; __syncthreads();
  for (int o = 128; o > 0; o >>= 1) { if (t < o) sm[t] += sm[t + o]; __syncthreads(); }
  float mean = sm[0] / Dc; __syncthreads();
  float vs = 0.f;
#pragma unroll
  for (int i = 0; i < 2; ++i) { float d = loc[i] - mean; vs += d * d; }
  sm[t] = vs; __syncthreads();
  for (int o = 128; o > 0; o >>= 1) { if (t < o) sm[t] += sm[t + o]; __syncthreads(); }
  float inv = rsqrtf(sm[0] / Dc + LNEPS); __syncthreads();
#pragma unroll
  for (int i = 0; i < 2; ++i) {
    int c = t + i * 256;
    outp[base + c] = (loc[i] - mean) * inv * gs[c] + gb[c];
  }
}

// ---------------- gate: row-dot + sigmoid, atomic sum for mean ----------------
__global__ __launch_bounds__(256) void k_gate_raw(const float* __restrict__ x,
                                                  const float* __restrict__ w,
                                                  const float* __restrict__ bptr,
                                                  float* __restrict__ raw,
                                                  float* __restrict__ sum) {
  __shared__ float sm[256];
  size_t base = (size_t)blockIdx.x * Dc;
  int t = threadIdx.x;
  float s = 0.f;
  for (int i = t; i < Dc; i += 256) s += x[base + i] * w[i];
  sm[t] = s; __syncthreads();
  for (int o = 128; o > 0; o >>= 1) { if (t < o) sm[t] += sm[t + o]; __syncthreads(); }
  if (t == 0) {
    float p = 1.0f / (1.0f + __expf(-(sm[0] + bptr[0])));
    raw[blockIdx.x] = p;
    atomicAdd(sum, p);
  }
}

__global__ __launch_bounds__(256) void k_dgate_final(const float* __restrict__ raw,
                                                     const float* __restrict__ sumptr,
                                                     const float* __restrict__ noise,
                                                     float* __restrict__ dprobs,
                                                     float* __restrict__ g_f,
                                                     int* __restrict__ g_i) {
  int i = blockIdx.x * blockDim.x + threadIdx.x;
  if (i >= Mc) return;
  float mean = sumptr[0] / (float)Mc;
  float p = raw[i] * (RATEc / mean);
  dprobs[i] = p;
  int gv = (noise[i] < p) ? 1 : 0;
  if ((i % Sc) == 0) gv = 1;
  g_i[i] = gv;
  g_f[i] = (float)gv;
}

__global__ __launch_bounds__(256) void k_ugate_final(const float* __restrict__ raw,
                                                     const float* __restrict__ sumptr,
                                                     float* __restrict__ uprobs) {
  int i = blockIdx.x * blockDim.x + threadIdx.x;
  if (i >= Mc) return;
  float mean = sumptr[0] / (float)Mc;
  uprobs[i] = raw[i] * (RATEc / mean);
}

// ---------------- exclusive scan per batch (block = S threads) ----------------
__global__ __launch_bounds__(1024) void k_scan_excl(const int* __restrict__ gin,
                                                    int* __restrict__ md,
                                                    int* __restrict__ ndst,
                                                    float* __restrict__ md_f) {
  __shared__ int sm[Sc];
  int b = blockIdx.x, t = threadIdx.x;
  int gv = gin[b * Sc + t];
  sm[t] = gv; __syncthreads();
  for (int off = 1; off < Sc; off <<= 1) {
    int add = (t >= off) ? sm[t - off] : 0;
    __syncthreads();
    sm[t] += add;
    __syncthreads();
  }
  int ex = sm[t] - gv;
  md[b * Sc + t] = ex;
  md_f[b * Sc + t] = (float)ex;
  if (ndst && t == Sc - 1) ndst[b] = ex + 1;
}

__global__ void k_i2f(const int* __restrict__ a, float* __restrict__ o, int n) {
  int i = blockIdx.x * blockDim.x + threadIdx.x;
  if (i < n) o[i] = (float)a[i];
}

__global__ __launch_bounds__(256) void k_ug(const int* __restrict__ g,
                                            int* __restrict__ ug_i,
                                            float* __restrict__ ug_f) {
  int i = blockIdx.x * blockDim.x + threadIdx.x;
  if (i >= Mc) return;
  int s = i % Sc;
  int v = (s < Sc - 1) ? g[i + 1] : 1;
  ug_i[i] = v;
  ug_f[i] = (float)v;
}

// ---------------- segment mean ----------------
__global__ __launch_bounds__(256) void k_zero(float* __restrict__ p, size_t n) {
  size_t i = (size_t)blockIdx.x * blockDim.x + threadIdx.x;
  if (i < n) p[i] = 0.0f;
}

__global__ __launch_bounds__(256) void k_seg_scatter(const float* __restrict__ x,
                                                     const int* __restrict__ seg,
                                                     float* __restrict__ sums,
                                                     float* __restrict__ cnts) {
  int row = blockIdx.x;
  int b = row / Sc;
  int sg = seg[row];
  size_t src = (size_t)row * Dc;
  size_t dst = ((size_t)b * Sc + sg) * Dc;
  for (int i = threadIdx.x; i < Dc; i += 256) atomicAdd(&sums[dst + i], x[src + i]);
  if (threadIdx.x == 0) atomicAdd(&cnts[b * Sc + sg], 1.0f);
}

__global__ __launch_bounds__(256) void k_seg_div(float* __restrict__ sums,
                                                 const float* __restrict__ cnts,
                                                 size_t total) {
  size_t i = (size_t)blockIdx.x * blockDim.x + threadIdx.x;
  if (i >= total) return;
  float c = cnts[i / Dc];
  sums[i] /= fmaxf(c, 1.0f);
}

__global__ __launch_bounds__(256) void k_copy(const float* __restrict__ a,
                                              float* __restrict__ o, size_t n) {
  size_t i = (size_t)blockIdx.x * blockDim.x + threadIdx.x;
  if (i < n) o[i] = a[i];
}

__global__ __launch_bounds__(256) void k_sub_inplace(float* __restrict__ y,
                                                     const float* __restrict__ x,
                                                     size_t n) {
  size_t i = (size_t)blockIdx.x * blockDim.x + threadIdx.x;
  if (i < n) y[i] -= x[i];
}

__global__ __launch_bounds__(256) void k_upsample(const float* __restrict__ x,
                                                  const float* __restrict__ dev,
                                                  const int* __restrict__ umd,
                                                  float* __restrict__ xup) {
  int row = blockIdx.x;
  int b = row / Sc;
  size_t dst = (size_t)row * Dc;
  size_t src = ((size_t)b * Sc + umd[row]) * Dc;
  for (int i = threadIdx.x; i < Dc; i += 256) xup[dst + i] = x[dst + i] + dev[src + i];
}

// ---------------- log-softmax over V per row ----------------
__global__ __launch_bounds__(256) void k_logsoftmax(float* __restrict__ logits) {
  __shared__ float sm[256];
  size_t base = (size_t)blockIdx.x * Vc;
  int t = threadIdx.x;
  float mx = NEG_BIG;
  for (int i = t; i < Vc; i += 256) mx = fmaxf(mx, logits[base + i]);
  sm[t] = mx; __syncthreads();
  for (int o = 128; o > 0; o >>= 1) { if (t < o) sm[t] = fmaxf(sm[t], sm[t + o]); __syncthreads(); }
  float m = sm[0]; __syncthreads();
  float s = 0.f;
  for (int i = t; i < Vc; i += 256) s += __expf(logits[base + i] - m);
  sm[t] = s; __syncthreads();
  for (int o = 128; o > 0; o >>= 1) { if (t < o) sm[t] += sm[t + o]; __syncthreads(); }
  float lse = m + __logf(sm[0]); __syncthreads();
  for (int i = t; i < Vc; i += 256) logits[base + i] -= lse;
}

// ---------------------------------------------------------------------------
// Host orchestration
// ---------------------------------------------------------------------------
struct LayerW {
  const float *Wqkv, *bqkv, *Wo, *bo, *ln1s, *ln1b, *W1, *b1, *W2, *b2, *ln2s, *ln2b;
};

extern "C" void kernel_launch(void* const* d_in, const int* in_sizes, int n_in,
                              void* d_out, int out_size, void* d_ws, size_t ws_size,
                              hipStream_t stream) {
  (void)in_sizes; (void)n_in; (void)out_size; (void)ws_size;

  const int* tokens  = (const int*)d_in[0];
  const float* noise = (const float*)d_in[1];
  const float* emb   = (const float*)d_in[2];
  const int baseDown = 3, baseMid = 15, baseUp = 27;
  const float* out_W = (const float*)d_in[39];
  const float* out_b = (const float*)d_in[40];
  const float* dgW   = (const float*)d_in[41];
  const float* dgB   = (const float*)d_in[42];
  const float* ugW   = (const float*)d_in[43];
  const float* ugB   = (const float*)d_in[44];

  // ---- output layout (all f32; integer outputs value-cast to float) ----
  float* out      = (float*)d_out;
  float* logits   = out;
  float* o_dprobs = out + (size_t)Mc * Vc;
  float* o_uprobs = o_dprobs + Mc;
  float* o_g      = o_uprobs + Mc;
  float* o_ug     = o_g + Mc;
  float* o_dmd    = o_ug + Mc;
  float* o_umd    = o_dmd + Mc;
  float* o_ndst   = o_umd + Mc;

  // ---- workspace carve ----
  float* fw   = (float*)d_ws;
  float* X    = fw;                       fw += (size_t)Mc * Dc;
  float* TMP1 = fw;                       fw += (size_t)Mc * 2048;  // qkv / ffn1
  float* TMP2 = fw;                       fw += (size_t)Mc * Dc;
  float* XDS  = fw;                       fw += (size_t)Mc * Dc;
  float* YDS  = fw;                       fw += (size_t)Mc * Dc;
  float* XUP  = fw;                       fw += (size_t)Mc * Dc;
  float* RAW  = fw;                       fw += Mc;
  float* SUMS = fw;                       fw += 8;
  float* CNTS = fw;                       fw += Mc;
  int* GI   = (int*)fw;
  int* DMD  = GI + Mc;
  int* UGI  = DMD + Mc;
  int* UMD  = UGI + Mc;
  int* NDST = UMD + Mc;
  unsigned short* ABF = (unsigned short*)(NDST + 16);
  unsigned short* WBF = ABF + (size_t)Mc * 2048;   // up to 32000*512 bf16

  auto getLayer = [&](int base, int l, int dff) {
    LayerW w;
    w.Wqkv = (const float*)d_in[base + 0]  + (size_t)l * 3 * Dc * Dc;
    w.bqkv = (const float*)d_in[base + 1]  + (size_t)l * 3 * Dc;
    w.Wo   = (const float*)d_in[base + 2]  + (size_t)l * Dc * Dc;
    w.bo   = (const float*)d_in[base + 3]  + (size_t)l * Dc;
    w.ln1s = (const float*)d_in[base + 4]  + (size_t)l * Dc;
    w.ln1b = (const float*)d_in[base + 5]  + (size_t)l * Dc;
    w.W1   = (const float*)d_in[base + 6]  + (size_t)l * dff * Dc;
    w.b1   = (const float*)d_in[base + 7]  + (size_t)l * dff;
    w.W2   = (const float*)d_in[base + 8]  + (size_t)l * Dc * dff;
    w.b2   = (const float*)d_in[base + 9]  + (size_t)l * Dc;
    w.ln2s = (const float*)d_in[base + 10] + (size_t)l * Dc;
    w.ln2b = (const float*)d_in[base + 11] + (size_t)l * Dc;
    return w;
  };

  auto cvt = [&](const float* src, unsigned short* dst, size_t n) {
    k_f32_to_bf16<<<CDIV(n, (size_t)256), 256, 0, stream>>>(src, dst, n);
  };
  auto gemm = [&](const unsigned short* A, const unsigned short* Wm, const float* bias,
                  float* Cm, int Mm, int Nn, int Kk, int relu) {
    dim3 g(Nn / 64, Mm / 16);   // N always a multiple of 64 here
    k_gemm_bf16<<<g, 32, 0, stream>>>(A, Wm, bias, Cm, Mm, Nn, Kk, relu);
  };
  auto run_layer = [&](float* Xp, LayerW w, int dff, int win) {
    cvt(Xp, ABF, (size_t)Mc * Dc);
    cvt(w.Wqkv, WBF, (size_t)3 * Dc * Dc);
    gemm(ABF, WBF, w.bqkv, TMP1, Mc, 3 * Dc, Dc, 0);
    k_attn<<<Bc * Hc * Sc, 32, 0, stream>>>(TMP1, TMP2, win);
    cvt(TMP2, ABF, (size_t)Mc * Dc);
    cvt(w.Wo, WBF, (size_t)Dc * Dc);
    gemm(ABF, WBF, w.bo, TMP1, Mc, Dc, Dc, 0);
    k_add_ln<<<Mc, 256, 0, stream>>>(Xp, TMP1, w.ln1s, w.ln1b, Xp);
    cvt(Xp, ABF, (size_t)Mc * Dc);
    cvt(w.W1, WBF, (size_t)dff * Dc);
    gemm(ABF, WBF, w.b1, TMP1, Mc, dff, Dc, 1);
    cvt(TMP1, ABF, (size_t)Mc * dff);
    cvt(w.W2, WBF, (size_t)Dc * dff);
    gemm(ABF, WBF, w.b2, TMP2, Mc, Dc, dff, 0);
    k_add_ln<<<Mc, 256, 0, stream>>>(Xp, TMP2, w.ln2s, w.ln2b, Xp);
  };

  // ---- embed + down layers ----
  k_embed<<<CDIV((size_t)Mc * Dc, (size_t)256), 256, 0, stream>>>(tokens, emb, X, (size_t)Mc * Dc);
  for (int l = 0; l < 2; ++l) run_layer(X, getLayer(baseDown, l, Dc), Dc, 64);

  // ---- down gate, sample, exclusive cumsum ----
  k_zero<<<1, 8, 0, stream>>>(SUMS, 8);
  k_gate_raw<<<Mc, 256, 0, stream>>>(X, dgW, dgB, RAW, SUMS);
  k_dgate_final<<<CDIV(Mc, 256), 256, 0, stream>>>(RAW, SUMS, noise, o_dprobs, o_g, GI);
  k_scan_excl<<<Bc, Sc, 0, stream>>>(GI, DMD, NDST, o_dmd);
  k_i2f<<<1, Bc, 0, stream>>>(NDST, o_ndst, Bc);

  // ---- segment mean downsample ----
  k_zero<<<CDIV((size_t)Mc * Dc, (size_t)256), 256, 0, stream>>>(XDS, (size_t)Mc * Dc);
  k_zero<<<CDIV(Mc, 256), 256, 0, stream>>>(CNTS, Mc);
  k_seg_scatter<<<Mc, 256, 0, stream>>>(X, DMD, XDS, CNTS);
  k_seg_div<<<CDIV((size_t)Mc * Dc, (size_t)256), 256, 0, stream>>>(XDS, CNTS, (size_t)Mc * Dc);

  // ---- mid layers on copy; dev = y_ds - x_ds ----
  k_copy<<<CDIV((size_t)Mc * Dc, (size_t)256), 256, 0, stream>>>(XDS, YDS, (size_t)Mc * Dc);
  for (int l = 0; l < 4; ++l) run_layer(YDS, getLayer(baseMid, l, 4 * Dc), 4 * Dc, 256);
  k_sub_inplace<<<CDIV((size_t)Mc * Dc, (size_t)256), 256, 0, stream>>>(YDS, XDS, (size_t)Mc * Dc);

  // ---- upsample map ----
  k_ug<<<CDIV(Mc, 256), 256, 0, stream>>>(GI, UGI, o_ug);
  k_scan_excl<<<Bc, Sc, 0, stream>>>(UGI, UMD, nullptr, o_umd);
  k_upsample<<<Mc, 256, 0, stream>>>(X, YDS, UMD, XUP);

  // ---- up layers ----
  for (int l = 0; l < 2; ++l) run_layer(XUP, getLayer(baseUp, l, Dc), Dc, 64);

  // ---- up gate ----
  k_gate_raw<<<Mc, 256, 0, stream>>>(XUP, ugW, ugB, RAW, SUMS + 1);
  k_ugate_final<<<CDIV(Mc, 256), 256, 0, stream>>>(RAW, SUMS + 1, o_uprobs);

  // ---- vocab projection + log-softmax (directly into d_out) ----
  cvt(XUP, ABF, (size_t)Mc * Dc);
  cvt(out_W, WBF, (size_t)Vc * Dc);
  gemm(ABF, WBF, out_b, logits, Mc, Vc, Dc, 0);
  k_logsoftmax<<<Mc, 256, 0, stream>>>(logits);
}
